// SimpleAttention_45449343926378
// MI455X (gfx1250) — compile-verified, matrix operand order checked by reading, not run on hardware
//
#include <hip/hip_runtime.h>
#include <stdint.h>

typedef __bf16 bf16;
typedef __attribute__((ext_vector_type(16))) __bf16 v16bf;
typedef __attribute__((ext_vector_type(8)))  __bf16 v8bf;
typedef __attribute__((ext_vector_type(8)))  float  v8f;
typedef __attribute__((ext_vector_type(4)))  float  v4f;

#define DEV static __device__ __forceinline__

constexpr int BB   = 8;       // batch
constexpr int SS   = 8192;    // sequence
constexpr int DM   = 256;     // d_model
constexpr int NH   = 8;       // heads
constexpr int GR   = 48;      // padded gram dim (34 -> 48)
constexpr int DPAD = 64;      // padded per-head feature dim (2 pos + 32 + zeros)
constexpr int KFC  = 512;     // fc K dim padded: 8 heads * 64
constexpr int NPART = 32;     // split-K partials for the gram
constexpr int WLDS = 40;      // LDS W-tile row stride in halves (bank-conflict pad)
constexpr float INV_S = 1.0f / 8192.0f;

DEV v8f v8f_zero() {
  v8f z = {0.f,0.f,0.f,0.f,0.f,0.f,0.f,0.f};
  return z;
}

// bf16 WMMA: D = A*B + C, fp32 accumulate
DEV v8f wmma_bf16(v16bf a, v16bf b, v8f c) {
  return __builtin_amdgcn_wmma_f32_16x16x32_bf16(false, a, false, b, (short)0, c, false, false);
}

// CDNA5 async global->LDS copy (VGLOBAL GLOBAL_LOAD_ASYNC_TO_LDS_B128, ASYNCcnt)
DEV void async_load_b128(unsigned ldsoff, unsigned long long gaddr) {
  asm volatile("global_load_async_to_lds_b128 %0, %1, off"
               :: "v"(ldsoff), "v"(gaddr)
               : "memory");
}
DEV void wait_async_zero() {
  asm volatile("s_wait_asynccnt 0x0" ::: "memory");
}

// Fragment loader (A or B side, identical lane pattern):
// caller passes rowptr already offset to (row = tile+lane%16, k = kbase + (lane/16)*8)
DEV v16bf frag_from_bf16(const bf16* rowptr) {
  v8bf a = *(const v8bf*)(rowptr);
  v8bf b = *(const v8bf*)(rowptr + 16);
  return __builtin_shufflevector(a, b, 0,1,2,3,4,5,6,7,8,9,10,11,12,13,14,15);
}

DEV v16bf frag_from_f32(const float* rowptr) {
  v4f a0 = *(const v4f*)(rowptr);
  v4f a1 = *(const v4f*)(rowptr + 4);
  v4f b0 = *(const v4f*)(rowptr + 16);
  v4f b1 = *(const v4f*)(rowptr + 20);
  v16bf f;
#pragma unroll
  for (int i = 0; i < 4; ++i) {
    f[i]      = (bf16)a0[i];
    f[4 + i]  = (bf16)a1[i];
    f[8 + i]  = (bf16)b0[i];
    f[12 + i] = (bf16)b1[i];
  }
  return f;
}

// ---------------------------------------------------------------------------
// prep: convert weights to bf16; build padded fc weight [256][512]
// ---------------------------------------------------------------------------
__global__ void prep_kernel(const float* wq, const float* wk, const float* wv,
                            const float* fcw,
                            bf16* WqB, bf16* WkB, bf16* WvB, bf16* WfcB) {
  int i = blockIdx.x * blockDim.x + threadIdx.x;   // 131072 threads
  if (i < DM * DM) {
    WqB[i] = (bf16)wq[i];
    WkB[i] = (bf16)wk[i];
    WvB[i] = (bf16)wv[i];
  }
  if (i < DM * KFC) {
    int o = i / KFC, k = i % KFC, h = k >> 6, c = k & 63;
    WfcB[i] = (c < 34) ? (bf16)fcw[o * 272 + h * 34 + c] : (bf16)0.0f;
  }
}

// ---------------------------------------------------------------------------
// fill pos columns/rows and zero pads (re-done every launch -> deterministic)
// ---------------------------------------------------------------------------
__global__ void fill_extras_kernel(const float* pos, bf16* Qh, bf16* KhT, bf16* VhT) {
  int t = blockIdx.x * blockDim.x + threadIdx.x;   // B*NH*S
  if (t >= BB * NH * SS) return;
  int s = t % SS;
  int bh = t / SS;
  int b = bh / NH;
  float p0 = pos[((size_t)b * SS + s) * 2 + 0];
  float p1 = pos[((size_t)b * SS + s) * 2 + 1];

  bf16* q = Qh + ((size_t)bh * SS + s) * DPAD;
  q[0] = (bf16)(p0 * INV_S);
  q[1] = (bf16)(p1 * INV_S);
#pragma unroll
  for (int c = 34; c < DPAD; ++c) q[c] = (bf16)0.0f;

  size_t tb = (size_t)bh * GR * SS + s;
  KhT[tb]      = (bf16)p0;  KhT[tb + SS] = (bf16)p1;
  VhT[tb]      = (bf16)p0;  VhT[tb + SS] = (bf16)p1;
#pragma unroll
  for (int r = 34; r < GR; ++r) {
    KhT[tb + (size_t)r * SS] = (bf16)0.0f;
    VhT[tb + (size_t)r * SS] = (bf16)0.0f;
  }
}

// ---------------------------------------------------------------------------
// projection GEMM: X[64 rows] @ W^T (K=256) via bf16 WMMA, fp32 accumulate.
// B(weight) fragments come from LDS, staged per k-step by the whole block via
// GLOBAL_LOAD_ASYNC_TO_LDS_B128 (one 20KB slice per block instead of per-wave
// global fragment reads -> ~13x less L2 traffic).
// LNORM=false: write Qh [B,H,S,64] (scaled by 1/S).
// LNORM=true : per-head layer-norm, LDS transpose, write [B,H,48,S] coalesced.
// Block = 128 threads (4 waves), each wave owns 16 M rows x all 256 features.
// ---------------------------------------------------------------------------
template <bool LNORM>
__global__ __launch_bounds__(128) void proj_kernel(
    const float* X, const bf16* Wb, const float* bias,
    const float* lnw, const float* lnb,
    bf16* Qh, bf16* OutT) {
  __shared__ bf16 lbufW[256 * WLDS];  // weight slice [256 rows][32 k + pad]
  __shared__ bf16 lbuf[256 * 72];     // [feature][s_local] transpose staging

  const int tid = threadIdx.x, wave = tid >> 5, lane = tid & 31;
  const int lr = lane & 15, kh = (lane >> 4) << 3, rhi = (lane >> 4) * 8;
  const int g0 = blockIdx.x * 64;           // global row base (= b*S + s0)
  const int b = g0 / SS, s0 = g0 % SS;
  const int grow = g0 + wave * 16 + lr;     // this lane's A row
  const unsigned ldsWbase = (unsigned)(uintptr_t)(&lbufW[0]);

  v8f acc[16];
#pragma unroll
  for (int i = 0; i < 16; ++i) acc[i] = v8f_zero();

  // prefetch all A fragments (fp32 -> bf16) so global A loads overlap staging
  const float* arow = X + (size_t)grow * DM;
  v16bf afs[8];
#pragma unroll
  for (int t = 0; t < 8; ++t) afs[t] = frag_from_f32(arow + t * 32 + kh);

  for (int ks = 0; ks < 8; ++ks) {
    const int kk = ks * 32;
    // async-stage W[:, kk..kk+32): 256 rows x 64B = 1024 b128 chunks / 128 thr
#pragma unroll
    for (int i = 0; i < 8; ++i) {
      int c = tid + i * 128;
      int row = c >> 2, part = c & 3;
      unsigned ldsoff = ldsWbase + (unsigned)((row * WLDS + part * 8) * 2);
      async_load_b128(ldsoff,
          (unsigned long long)(uintptr_t)(Wb + (size_t)row * DM + kk + part * 8));
    }
    wait_async_zero();
    __syncthreads();
#pragma unroll
    for (int nt = 0; nt < 16; ++nt) {
      v16bf bfr = frag_from_bf16(lbufW + (nt * 16 + lr) * WLDS + kh);
      acc[nt] = wmma_bf16(afs[ks], bfr, acc[nt]);
    }
    __syncthreads();   // protect LDS slice before next stage overwrites
  }

  if (!LNORM) {
    // Qh[b,h,s, 2+j] = (x + bias) / S   (row-major, half-wave-coalesced b16 stores)
#pragma unroll
    for (int nt = 0; nt < 16; ++nt) {
      int f = nt * 16 + lr, h = f >> 5, j = f & 31;
#pragma unroll
      for (int v = 0; v < 8; ++v) {
        int s = s0 + wave * 16 + v + rhi;
        float val = (acc[nt][v] + bias[f]) * INV_S;
        Qh[(((size_t)b * NH + h) * SS + s) * DPAD + 2 + j] = (bf16)val;
      }
    }
  } else {
    // per-(row,head) layernorm over 32 features: cross-lane reduce in 16-lane halves
#pragma unroll
    for (int h = 0; h < NH; ++h) {
      int t0 = 2 * h, t1 = 2 * h + 1;
      float w0 = lnw[h * 32 + lr],      w1 = lnw[h * 32 + 16 + lr];
      float c0 = lnb[h * 32 + lr],      c1 = lnb[h * 32 + 16 + lr];
      float e0 = bias[h * 32 + lr],     e1 = bias[h * 32 + 16 + lr];
#pragma unroll
      for (int v = 0; v < 8; ++v) {
        float x0 = acc[t0][v] + e0;
        float x1 = acc[t1][v] + e1;
        float s1 = x0 + x1;
        float s2 = x0 * x0 + x1 * x1;
        for (int m = 1; m < 16; m <<= 1) {   // stays inside each 16-lane half
          s1 += __shfl_xor(s1, m, 32);
          s2 += __shfl_xor(s2, m, 32);
        }
        float mean = s1 * (1.0f / 32.0f);
        float var  = s2 * (1.0f / 32.0f) - mean * mean;
        float rstd = rsqrtf(var + 1e-5f);
        float y0 = (x0 - mean) * rstd * w0 + c0;
        float y1 = (x1 - mean) * rstd * w1 + c1;
        int sl = wave * 16 + v + rhi;
        lbuf[(h * 32 + lr) * 72 + sl]      = (bf16)y0;
        lbuf[(h * 32 + 16 + lr) * 72 + sl] = (bf16)y1;
      }
    }
    __syncthreads();
    // transposed coalesced write: OutT[b,h, 2+j, s0..s0+63]
    for (int f = tid; f < 256; f += 128) {
      int h = f >> 5, j = f & 31;
      bf16* dst = OutT + (((size_t)b * NH + h) * GR + 2 + j) * SS + s0;
#pragma unroll
      for (int c = 0; c < 64; c += 8)
        *(v8bf*)(dst + c) = *(const v8bf*)(&lbuf[f * 72 + c]);
    }
  }
}

// ---------------------------------------------------------------------------
// gram: partial[bh][part][d][e] = sum_{s in chunk} KhT[d][s]*VhT[e][s]
// block = 128 threads (4 waves); each wave = one split-K partial, 3x3 tiles
// ---------------------------------------------------------------------------
__global__ __launch_bounds__(128) void gram_kernel(const bf16* KhT, const bf16* VhT,
                                                   float* partial) {
  const int bh = blockIdx.x;
  const int tid = threadIdx.x, wave = tid >> 5, lane = tid & 31;
  const int lr = lane & 15, kh = (lane >> 4) << 3, rhi = (lane >> 4) * 8;
  const int part = blockIdx.y * 4 + wave;
  const int sbeg = part * (SS / NPART);   // 256 s per partial

  v8f acc[9];
#pragma unroll
  for (int i = 0; i < 9; ++i) acc[i] = v8f_zero();

  const bf16* Kb = KhT + (size_t)bh * GR * SS;
  const bf16* Vb = VhT + (size_t)bh * GR * SS;

  for (int kk = 0; kk < SS / NPART; kk += 32) {
    int sk = sbeg + kk + kh;
    v16bf af[3], bfr[3];
#pragma unroll
    for (int mt = 0; mt < 3; ++mt)
      af[mt] = frag_from_bf16(Kb + (size_t)(mt * 16 + lr) * SS + sk);
#pragma unroll
    for (int nt = 0; nt < 3; ++nt)
      bfr[nt] = frag_from_bf16(Vb + (size_t)(nt * 16 + lr) * SS + sk);
#pragma unroll
    for (int mt = 0; mt < 3; ++mt)
#pragma unroll
      for (int nt = 0; nt < 3; ++nt)
        acc[mt * 3 + nt] = wmma_bf16(af[mt], bfr[nt], acc[mt * 3 + nt]);
  }

  float* pb = partial + ((size_t)bh * NPART + part) * GR * GR;
#pragma unroll
  for (int mt = 0; mt < 3; ++mt)
#pragma unroll
    for (int nt = 0; nt < 3; ++nt)
#pragma unroll
      for (int v = 0; v < 8; ++v)
        pb[(size_t)(mt * 16 + v + rhi) * GR + nt * 16 + lr] = acc[mt * 3 + nt][v];
}

// ---------------------------------------------------------------------------
// reduce 32 partials -> A^T bf16 [bh][e][d] (d padded to 64 with zeros)
// ---------------------------------------------------------------------------
__global__ void gram_reduce_kernel(const float* partial, bf16* ATb) {
  int i = blockIdx.x * blockDim.x + threadIdx.x;   // 64*48*64
  if (i >= BB * NH * GR * DPAD) return;
  int d = i % DPAD;
  int rest = i / DPAD;
  int e = rest % GR;
  int bh = rest / GR;
  float s = 0.0f;
  if (d < GR) {
    const float* pb = partial + (size_t)bh * NPART * GR * GR + (size_t)d * GR + e;
    for (int p = 0; p < NPART; ++p) s += pb[(size_t)p * GR * GR];
  }
  ATb[((size_t)bh * GR + e) * DPAD + d] = (bf16)s;
}

// ---------------------------------------------------------------------------
// attn: Attn[s][e] = Qh[s][:] @ A  — in place into the Qh buffer
// block = 256 threads (8 waves), 128 s rows per block, 3 N tiles + zero tile
// ---------------------------------------------------------------------------
__global__ __launch_bounds__(256) void attn_kernel(const bf16* ATb, bf16* QhAttn) {
  const int bh = blockIdx.x;
  const int s0 = blockIdx.y * 128;
  const int tid = threadIdx.x, wave = tid >> 5, lane = tid & 31;
  const int lr = lane & 15, kh = (lane >> 4) << 3, rhi = (lane >> 4) * 8;
  const int srow = s0 + wave * 16 + lr;
  const bf16* qrow = QhAttn + ((size_t)bh * SS + srow) * DPAD;

  v8f acc[3];
#pragma unroll
  for (int i = 0; i < 3; ++i) acc[i] = v8f_zero();

#pragma unroll
  for (int kk = 0; kk < DPAD; kk += 32) {
    v16bf af = frag_from_bf16(qrow + kk + kh);
#pragma unroll
    for (int nt = 0; nt < 3; ++nt) {
      v16bf bfr = frag_from_bf16(ATb + ((size_t)bh * GR + nt * 16 + lr) * DPAD + kk + kh);
      acc[nt] = wmma_bf16(af, bfr, acc[nt]);
    }
  }
  // all reads of this wave's rows are complete; store in place
#pragma unroll
  for (int nt = 0; nt < 3; ++nt)
#pragma unroll
    for (int v = 0; v < 8; ++v) {
      int s = s0 + wave * 16 + v + rhi;
      QhAttn[((size_t)bh * SS + s) * DPAD + nt * 16 + lr] = (bf16)acc[nt][v];
    }
#pragma unroll
  for (int v = 0; v < 8; ++v) {   // keep cols 48..63 zero for the fc K-loop
    int s = s0 + wave * 16 + v + rhi;
    QhAttn[((size_t)bh * SS + s) * DPAD + 48 + lr] = (bf16)0.0f;
  }
}

// ---------------------------------------------------------------------------
// fc: out[b,s,o] = sum_{h,c} Attn[b,h,s,c] * Wfc[o][h*64+c] + fc_b[o]
// block = 256 threads (8 waves), 128 M rows, K=512, N=256 (16 tiles)
// Weight slice async-staged to LDS per k-step, B fragments read via ds_load.
// ---------------------------------------------------------------------------
__global__ __launch_bounds__(256) void fc_kernel(const bf16* Attn, const bf16* WfcB,
                                                 const float* fcb, float* out) {
  __shared__ bf16 lbufW[256 * WLDS];
  const int tid = threadIdx.x, wave = tid >> 5, lane = tid & 31;
  const int lr = lane & 15, kh = (lane >> 4) << 3, rhi = (lane >> 4) * 8;
  const int g0 = blockIdx.x * 128;
  const int b = g0 / SS, s0 = g0 % SS;
  const int srow = s0 + wave * 16 + lr;
  const unsigned ldsWbase = (unsigned)(uintptr_t)(&lbufW[0]);

  v8f acc[16];
#pragma unroll
  for (int i = 0; i < 16; ++i) acc[i] = v8f_zero();

  for (int ks = 0; ks < 16; ++ks) {
    const int kk = ks * 32;
    // async-stage Wfc[:, kk..kk+32): 1024 b128 chunks / 256 threads
#pragma unroll
    for (int i = 0; i < 4; ++i) {
      int c = tid + i * 256;
      int row = c >> 2, part = c & 3;
      unsigned ldsoff = ldsWbase + (unsigned)((row * WLDS + part * 8) * 2);
      async_load_b128(ldsoff,
          (unsigned long long)(uintptr_t)(WfcB + (size_t)row * KFC + kk + part * 8));
    }
    int h = kk >> 6, kc = kk & 63;
    v16bf af = frag_from_bf16(Attn + (((size_t)b * NH + h) * SS + srow) * DPAD + kc + kh);
    wait_async_zero();
    __syncthreads();
#pragma unroll
    for (int nt = 0; nt < 16; ++nt) {
      v16bf bfr = frag_from_bf16(lbufW + (nt * 16 + lr) * WLDS + kh);
      acc[nt] = wmma_bf16(af, bfr, acc[nt]);
    }
    __syncthreads();
  }
#pragma unroll
  for (int nt = 0; nt < 16; ++nt) {
    int o = nt * 16 + lr;
    float bias = fcb[o];
#pragma unroll
    for (int v = 0; v < 8; ++v) {
      int g = g0 + wave * 16 + v + rhi;
      out[(size_t)g * DM + o] = acc[nt][v] + bias;
    }
  }
}

// ---------------------------------------------------------------------------
extern "C" void kernel_launch(void* const* d_in, const int* in_sizes, int n_in,
                              void* d_out, int out_size, void* d_ws, size_t ws_size,
                              hipStream_t stream) {
  const float* Q      = (const float*)d_in[0];
  const float* K      = (const float*)d_in[1];
  const float* V      = (const float*)d_in[2];
  const float* pos    = (const float*)d_in[3];
  const float* wq     = (const float*)d_in[4];
  const float* wk     = (const float*)d_in[5];
  const float* wv     = (const float*)d_in[6];
  const float* bq     = (const float*)d_in[7];
  const float* bk     = (const float*)d_in[8];
  const float* bv     = (const float*)d_in[9];
  const float* lnwK   = (const float*)d_in[10];
  const float* lnbK   = (const float*)d_in[11];
  const float* lnwV   = (const float*)d_in[12];
  const float* lnbV   = (const float*)d_in[13];
  const float* fcw    = (const float*)d_in[14];
  const float* fcb    = (const float*)d_in[15];
  float* out = (float*)d_out;

  char* w = (char*)d_ws;
  bf16* WqB  = (bf16*)w;  w += (size_t)DM * DM * 2;
  bf16* WkB  = (bf16*)w;  w += (size_t)DM * DM * 2;
  bf16* WvB  = (bf16*)w;  w += (size_t)DM * DM * 2;
  bf16* WfcB = (bf16*)w;  w += (size_t)DM * KFC * 2;
  bf16* Qh   = (bf16*)w;  w += (size_t)BB * NH * SS * DPAD * 2;   // also holds Attn
  bf16* KhT  = (bf16*)w;  w += (size_t)BB * NH * GR * SS * 2;
  bf16* VhT  = (bf16*)w;  w += (size_t)BB * NH * GR * SS * 2;
  float* partial = (float*)w; w += (size_t)BB * NH * NPART * GR * GR * 4;
  bf16* ATb  = (bf16*)w;  w += (size_t)BB * NH * GR * DPAD * 2;
  (void)ws_size; (void)in_sizes; (void)n_in; (void)out_size;

  // 1) weights -> bf16 (incl. padded fc weight)
  prep_kernel<<<(DM * KFC) / 256, 256, 0, stream>>>(wq, wk, wv, fcw, WqB, WkB, WvB, WfcB);
  // 2) pos columns/rows + zero pads (must precede gram/attn reads)
  fill_extras_kernel<<<(BB * NH * SS) / 256, 256, 0, stream>>>(pos, Qh, KhT, VhT);
  // 3) projections
  proj_kernel<false><<<(BB * SS) / 64, 128, 0, stream>>>(Q, WqB, bq, nullptr, nullptr, Qh, nullptr);
  proj_kernel<true ><<<(BB * SS) / 64, 128, 0, stream>>>(K, WkB, bk, lnwK, lnbK, nullptr, KhT);
  proj_kernel<true ><<<(BB * SS) / 64, 128, 0, stream>>>(V, WvB, bv, lnwV, lnbV, nullptr, VhT);
  // 4) gram (split-K) + reduce/transpose to bf16
  gram_kernel<<<dim3(BB * NH, NPART / 4), 128, 0, stream>>>(KhT, VhT, partial);
  gram_reduce_kernel<<<(BB * NH * GR * DPAD + 255) / 256, 256, 0, stream>>>(partial, ATb);
  // 5) attn (in place into Qh buffer)
  attn_kernel<<<dim3(BB * NH, SS / 128), 256, 0, stream>>>(ATb, Qh);
  // 6) final fc
  fc_kernel<<<(BB * SS) / 128, 256, 0, stream>>>(Qh, WfcB, fcb, out);
}
